// ASR_4707284156838
// MI455X (gfx1250) — 16-bit and 8-bit matrix operand order repaired and checked by reading, compiled, not run on hardware
//
#include <hip/hip_runtime.h>
#include <hip/hip_bf16.h>

typedef __attribute__((ext_vector_type(16))) _Float16 v16h;
typedef __attribute__((ext_vector_type(8)))  float    v8f;
typedef __attribute__((ext_vector_type(8)))  int      v8i;
typedef __attribute__((ext_vector_type(16))) int      v16i;
typedef __attribute__((ext_vector_type(4)))  int      v4i;
typedef __attribute__((ext_vector_type(4)))  unsigned int v4u;
typedef unsigned int u32;

#if defined(__has_builtin)
#  if __has_builtin(__builtin_amdgcn_tensor_load_to_lds) && __has_builtin(__builtin_amdgcn_s_wait_tensorcnt)
#    define HAVE_TDM 1
#  endif
#  if __has_builtin(__builtin_amdgcn_global_load_async_to_lds_b128) && __has_builtin(__builtin_amdgcn_s_wait_asynccnt)
#    define HAVE_ASYNC 1
#  endif
#endif
#ifndef HAVE_TDM
#define HAVE_TDM 0
#endif
#ifndef HAVE_ASYNC
#define HAVE_ASYNC 0
#endif

namespace {
constexpr int    kT    = 1600;
constexpr int    kB    = 64;
constexpr int    kF    = 13;
constexpr int    kH    = 256;
constexpr int    kG    = 1024;
constexpr int    kV    = 34;
constexpr long long kRows = (long long)kT * kB;

constexpr size_t WHH8_MAT  = 64ull * 4 * 32 * 32;
constexpr size_t WHH8_OFF  = 0;
constexpr size_t WIH16_MAT = 64ull * 16 * 32 * 16;
constexpr size_t WIH16_OFF = 2ull * 1024 * 1024;
constexpr size_t B0_OFF    = 6ull * 1024 * 1024;
constexpr size_t B12_OFF   = B0_OFF + 2ull * 1024 * 4;
constexpr size_t PRE_OFF   = 8ull * 1024 * 1024;
constexpr size_t YA_OFF    = PRE_OFF + 2ull * 102400 * 1024 * 4;
constexpr size_t YB_OFF    = YA_OFF + (size_t)102400 * 512 * 2;
constexpr size_t SMEM_SCAN = 262144 + 4096 + 32768;
constexpr size_t SMEM_GEMM = 64ull * 512 * 2;
constexpr size_t SMEM_PROJ = 34ull*512*4 + 8ull*512*4 + 64*4;
}

__device__ __forceinline__ unsigned char f32_to_e4m3(float f) {
    union { float f; u32 u; } v; v.f = f;
    unsigned char s = (unsigned char)((v.u >> 24) & 0x80u);
    u32 be = (v.u >> 23) & 0xFFu;
    if (be == 0) return s;
    int e = (int)be - 127;
    float af = fabsf(f);
    int ee = e + 7;
    if (ee >= 1) {
        u32 m = v.u & 0x7FFFFFu;
        u32 mi = (m + (1u << 19)) >> 20;
        if (mi == 8u) { mi = 0u; ee++; }
        if (ee > 15 || (ee == 15 && mi > 6u)) return (unsigned char)(s | 0x7Eu);
        return (unsigned char)(s | ((u32)ee << 3) | mi);
    } else {
        float q = af * 512.f;
        int mi = (int)(q + 0.5f);
        if (mi > 7) return (unsigned char)(s | 0x08u);
        return (unsigned char)(s | (u32)mi);
    }
}

__global__ __launch_bounds__(256) void prep_bias(
        const float* __restrict__ bi0, const float* __restrict__ bh0,
        const float* __restrict__ bi12, const float* __restrict__ bh12,
        float* __restrict__ b0, float* __restrict__ b12) {
    unsigned i = blockIdx.x * 256u + threadIdx.x;
    if (i < 2048u) b0[i] = bi0[i] + bh0[i];
    else if (i < 2048u + 4096u) { unsigned j = i - 2048u; b12[j] = bi12[j] + bh12[j]; }
}

__global__ __launch_bounds__(256) void pack_whh_fp8(
        const float* __restrict__ whh0, const float* __restrict__ whh12,
        unsigned char* __restrict__ dst) {
    unsigned idx = blockIdx.x * 256u + threadIdx.x;
    unsigned mat  = idx >> 18;
    unsigned rem  = idx & 262143u;
    unsigned nt   = rem >> 12;
    unsigned kt   = (rem >> 10) & 3u;
    unsigned lane = (rem >> 5) & 31u;
    unsigned p    = rem & 31u;
    unsigned n = nt * 16u + (lane & 15u);
    unsigned k = kt * 64u + ((p >> 4) & 1u) * 32u + (lane >> 4) * 16u + ((p >> 2) & 3u) * 4u + (p & 3u);
    unsigned L = mat >> 1, dir = mat & 1u;
    float w = (L == 0u) ? whh0 [((size_t)dir * kG + n) * kH + k]
                        : whh12[(((size_t)(L - 1u) * 2 + dir) * kG + n) * kH + k];
    dst[idx] = f32_to_e4m3(w);
}

__global__ __launch_bounds__(256) void pack_wih_f16(
        const float* __restrict__ wih12, _Float16* __restrict__ dst) {
    unsigned idx = blockIdx.x * 256u + threadIdx.x;
    unsigned mat  = idx >> 19;
    unsigned rem  = idx & 524287u;
    unsigned nt   = rem >> 13;
    unsigned kt   = (rem >> 9) & 15u;
    unsigned lane = (rem >> 4) & 31u;
    unsigned e    = rem & 15u;
    unsigned n = nt * 16u + (lane & 15u);
    unsigned k = kt * 32u + (lane >> 4) * 16u + e;
    dst[idx] = (_Float16)wih12[((size_t)mat * kG + n) * (2 * kH) + k];
}

__global__ __launch_bounds__(256) void pre_l0(
        const float* __restrict__ x, const float* __restrict__ wih0,
        const float* __restrict__ b0, float* __restrict__ pre) {
    size_t idx = (size_t)blockIdx.x * 256 + threadIdx.x;
    unsigned n   = (unsigned)(idx & 1023u);
    size_t   rr  = idx >> 10;
    unsigned dir = (unsigned)(rr / (size_t)kRows);
    size_t   row = rr % (size_t)kRows;
    const float* w  = wih0 + ((size_t)dir * kG + n) * kF;
    const float* xr = x + row * kF;
    float s = b0[dir * kG + n];
#pragma unroll
    for (int i = 0; i < kF; ++i) s += xr[i] * w[i];
    pre[idx] = s;
}

__global__ __launch_bounds__(256) void pre_gemm(
        const _Float16* __restrict__ Yin,
        const _Float16* __restrict__ wpack,
        const float* __restrict__ bias,
        float* __restrict__ pre) {
    extern __shared__ unsigned char smem_g[];
    _Float16* ytile = (_Float16*)smem_g;
    const int tid = threadIdx.x, lane = tid & 31, wave = tid >> 5;

    {
        const uint4* src = (const uint4*)(Yin + (size_t)blockIdx.x * 64 * 512);
        uint4* dl = (uint4*)ytile;
#if HAVE_ASYNC
        typedef __attribute__((address_space(1))) v4i* v4i_gp;
        typedef __attribute__((address_space(3))) v4i* v4i_lp;
#pragma unroll
        for (int i = 0; i < 16; ++i)
            __builtin_amdgcn_global_load_async_to_lds_b128(
                (v4i_gp)(uint4*)(src + tid + i * 256),
                (v4i_lp)(dl + tid + i * 256), 0, 0);
        __builtin_amdgcn_s_wait_asynccnt(0);
#else
#pragma unroll
        for (int i = 0; i < 16; ++i) dl[tid + i * 256] = src[tid + i * 256];
#endif
    }
    __syncthreads();

    const int dir = blockIdx.z;
    const int nt  = blockIdx.y * 8 + wave;
    const _Float16* bbase = wpack + (((size_t)dir * 64 + nt) * 16) * 32 * 16;
    const int arow = lane & 15, ahalf = lane >> 4;

    v8f acc0 = {}, acc1 = {}, acc2 = {}, acc3 = {};
    for (int kt = 0; kt < 16; ++kt) {
        v16h Bf;
        {
            const _Float16* bp = bbase + ((size_t)kt * 32 + lane) * 16;
#pragma unroll
            for (int e = 0; e < 16; ++e) Bf[e] = bp[e];
        }
#pragma unroll
        for (int mr = 0; mr < 4; ++mr) {
            const _Float16* ap = ytile + ((mr * 16 + arow) * 512 + kt * 32 + ahalf * 16);
            v16h A;
#pragma unroll
            for (int e = 0; e < 16; ++e) A[e] = ap[e];
            v8f& acc = mr == 0 ? acc0 : mr == 1 ? acc1 : mr == 2 ? acc2 : acc3;
            acc = __builtin_amdgcn_wmma_f32_16x16x32_f16(false, A, false, Bf, (short)0, acc, false, false);
        }
    }
    const int n = nt * 16 + (lane & 15);
    const float bv = bias[dir * kG + n];
#pragma unroll
    for (int mr = 0; mr < 4; ++mr) {
        v8f acc = mr == 0 ? acc0 : mr == 1 ? acc1 : mr == 2 ? acc2 : acc3;
        size_t rowb = (size_t)blockIdx.x * 64 + mr * 16;
        float* dp = pre + ((size_t)dir * kRows + rowb) * kG + n;
#pragma unroll
        for (int r = 0; r < 8; ++r) dp[(size_t)(r + ahalf * 8) * kG] = acc[r] + bv;
    }
}

__global__ __launch_bounds__(256) void lstm_scan(
        const unsigned char* __restrict__ wfp8_layer,
        const float* __restrict__ pre,
        _Float16* __restrict__ Yout) {
    extern __shared__ unsigned char smem[];
    u32*           wlds = (u32*)smem;
    unsigned char* hlds = smem + 262144;
    float*         glds = (float*)(smem + 266240);

    const int tid = threadIdx.x, lane = tid & 31, wave = tid >> 5;
    const int bch = blockIdx.x, dir = blockIdx.y;
    const int am = tid & 15, ajb = (tid >> 4) * 16;

#if HAVE_TDM
    if (tid == 0) {
        unsigned long long ga = (unsigned long long)(wfp8_layer + (size_t)dir * WHH8_MAT);
        v4u g0;
        g0[0] = 1u;
        g0[1] = 0u;
        g0[2] = (u32)ga;
        g0[3] = (u32)((ga >> 32) & 0x1FFFFFFu) | (2u << 30);
        v8i g1;
        g1[0] = 0x30000;
        g1[1] = (int)0x80000000u;
        g1[2] = 0x00010000;
        g1[3] = (int)0x80000000u;
        g1[4] = 0x00000001;
        g1[5] = 32768;
        g1[6] = 0; g1[7] = 0;
        v4i z4 = {0, 0, 0, 0};
        v8i z8 = {0, 0, 0, 0, 0, 0, 0, 0};
        __builtin_amdgcn_tensor_load_to_lds(g0, g1, z4, z4, z8, 0);
    }
    __builtin_amdgcn_s_wait_tensorcnt(0);
#else
    {
        const uint4* src = (const uint4*)(wfp8_layer + (size_t)dir * WHH8_MAT);
        uint4* dl = (uint4*)wlds;
#pragma unroll
        for (int i = 0; i < 64; ++i) dl[tid + i * 256] = src[tid + i * 256];
    }
#endif
    for (int i = tid; i < 1024; i += 256) ((u32*)hlds)[i] = 0u;

    float c[16], pi[16], pf[16];
#pragma unroll
    for (int u = 0; u < 16; ++u) c[u] = 0.f;
    __syncthreads();

    for (int t = 0; t < kT; ++t) {
        const int tt = dir ? (kT - 1 - t) : t;
        const float* prow = pre + ((size_t)dir * kRows + (size_t)tt * kB + bch * 16) * kG;

        if (t + 1 < kT) {
            const int tn = dir ? (kT - 2 - t) : (t + 1);
            const float* pn = pre + ((size_t)dir * kRows + (size_t)tn * kB + bch * 16) * kG
                              + (size_t)tid * 64;
            __builtin_prefetch(pn, 0, 1);
            __builtin_prefetch(pn + 32, 0, 1);
        }

#pragma unroll 1
        for (int half = 0; half < 2; ++half) {
#pragma unroll
            for (int q = 0; q < 4; ++q) {
                const int nt = half * 32 + wave * 4 + q;
                v8f acc = {};
#pragma unroll
                for (int k2 = 0; k2 < 2; ++k2) {
                    const unsigned char* hrow = hlds + (lane & 15) * 256 + k2 * 128 + ((lane >> 4) << 4);
                    v16i A;
#pragma unroll
                    for (int v = 0; v < 16; ++v) {
                        const int vv = v & 7;
                        A[v] = *(const int*)(hrow + ((v >> 3) << 6) + ((vv >> 2) << 5) + ((vv & 3) << 2));
                    }
                    const u32* bp = wlds + (((size_t)nt * 4 + k2 * 2) * 32 + lane) * 8;
                    v16i Bm;
#pragma unroll
                    for (int v = 0; v < 8; ++v) { Bm[v] = (int)bp[v]; Bm[v + 8] = (int)bp[v + 256]; }
                    acc = __builtin_amdgcn_wmma_f32_16x16x128_fp8_fp8(A, Bm, (short)0, acc, false, false);
                }
                const int nl = (nt - half * 32) * 16 + (lane & 15);
#pragma unroll
                for (int r = 0; r < 8; ++r) glds[(r + ((lane >> 4) << 3)) * 512 + nl] = acc[r];
            }
            __syncthreads();

            if (half == 0) {
#pragma unroll
                for (int u = 0; u < 16; ++u) {
                    const int j = ajb + u;
                    pi[u] = glds[am * 512 + j]       + prow[(size_t)am * kG + j];
                    pf[u] = glds[am * 512 + 256 + j] + prow[(size_t)am * kG + 256 + j];
                }
                __syncthreads();
            } else {
                union { unsigned char b[16]; uint4 q4; } hb;
                _Float16* yp = Yout + ((size_t)tt * kB + bch * 16 + am) * (2 * kH) + dir * kH + ajb;
#pragma unroll
                for (int u = 0; u < 16; ++u) {
                    const int j = ajb + u;
                    float gg = glds[am * 512 + j]       + prow[(size_t)am * kG + 512 + j];
                    float go = glds[am * 512 + 256 + j] + prow[(size_t)am * kG + 768 + j];
                    float si = 1.f / (1.f + __expf(-pi[u]));
                    float sf = 1.f / (1.f + __expf(-pf[u]));
                    float so = 1.f / (1.f + __expf(-go));
                    float cc = sf * c[u] + si * tanhf(gg);
                    c[u] = cc;
                    float hv = so * tanhf(cc);
                    hb.b[u] = f32_to_e4m3(hv);
                    yp[u] = (_Float16)hv;
                }
                *(uint4*)(hlds + am * 256 + ajb) = hb.q4;
                __syncthreads();
            }
        }
    }
}

__global__ __launch_bounds__(256) void proj_lsm(
        const _Float16* __restrict__ Y, const float* __restrict__ wout,
        const float* __restrict__ bout, float* __restrict__ out) {
    extern __shared__ unsigned char smem_p[];
    float* wl = (float*)smem_p;
    float* yl = (float*)(smem_p + 34 * 512 * 4);
    float* lg = (float*)(smem_p + 34 * 512 * 4 + 8 * 512 * 4);
    const int tid = threadIdx.x;

    for (int i = tid; i < 34 * 512; i += 256) wl[i] = wout[i];
    const _Float16* yb = Y + (size_t)blockIdx.x * 8 * 512;
    for (int i = tid; i < 8 * 512; i += 256) yl[i] = (float)yb[i];
    __syncthreads();

    for (int r = 0; r < 8; ++r) {
        if (tid < kV) {
            float s = bout[tid];
            const float* w = wl + tid * 512;
            const float* y = yl + r * 512;
            for (int k = 0; k < 512; ++k) s += y[k] * w[k];
            lg[tid] = s;
        }
        __syncthreads();
        if (tid == 0) {
            float mx = lg[0];
            for (int v = 1; v < kV; ++v) mx = fmaxf(mx, lg[v]);
            float se = 0.f;
            for (int v = 0; v < kV; ++v) se += __expf(lg[v] - mx);
            lg[kV] = mx + __logf(se);
        }
        __syncthreads();
        if (tid < kV) out[((size_t)blockIdx.x * 8 + r) * kV + tid] = lg[tid] - lg[kV];
        __syncthreads();
    }
}

extern "C" void kernel_launch(void* const* d_in, const int* in_sizes, int n_in,
                              void* d_out, int out_size, void* d_ws, size_t ws_size,
                              hipStream_t stream) {
    const float* x     = (const float*)d_in[0];
    const float* wih0  = (const float*)d_in[1];
    const float* whh0  = (const float*)d_in[2];
    const float* bih0  = (const float*)d_in[3];
    const float* bhh0  = (const float*)d_in[4];
    const float* wih12 = (const float*)d_in[5];
    const float* whh12 = (const float*)d_in[6];
    const float* bih12 = (const float*)d_in[7];
    const float* bhh12 = (const float*)d_in[8];
    const float* wout  = (const float*)d_in[9];
    const float* bout  = (const float*)d_in[10];

    unsigned char* ws = (unsigned char*)d_ws;
    unsigned char* whh8  = ws + WHH8_OFF;
    _Float16*      wih16 = (_Float16*)(ws + WIH16_OFF);
    float*         b0    = (float*)(ws + B0_OFF);
    float*         b12   = (float*)(ws + B12_OFF);
    float*         pre   = (float*)(ws + PRE_OFF);
    _Float16*      Ya    = (_Float16*)(ws + YA_OFF);
    _Float16*      Yb    = (_Float16*)(ws + YB_OFF);

    prep_bias   <<<24, 256, 0, stream>>>(bih0, bhh0, bih12, bhh12, b0, b12);
    pack_whh_fp8<<<6144, 256, 0, stream>>>(whh0, whh12, whh8);
    pack_wih_f16<<<8192, 256, 0, stream>>>(wih12, wih16);

    pre_l0   <<<819200, 256, 0, stream>>>(x, wih0, b0, pre);
    lstm_scan<<<dim3(4, 2), 256, SMEM_SCAN, stream>>>(whh8 + 0 * 2 * WHH8_MAT, pre, Ya);
    pre_gemm <<<dim3(1600, 8, 2), 256, SMEM_GEMM, stream>>>(Ya, wih16 + 0 * 2 * WIH16_MAT, b12, pre);
    lstm_scan<<<dim3(4, 2), 256, SMEM_SCAN, stream>>>(whh8 + 1 * 2 * WHH8_MAT, pre, Yb);
    pre_gemm <<<dim3(1600, 8, 2), 256, SMEM_GEMM, stream>>>(Yb, wih16 + 1 * 2 * WIH16_MAT, b12 + 2048, pre);
    lstm_scan<<<dim3(4, 2), 256, SMEM_SCAN, stream>>>(whh8 + 2 * 2 * WHH8_MAT, pre, Ya);
    proj_lsm <<<12800, 256, SMEM_PROJ, stream>>>(Ya, wout, bout, (float*)d_out);

    (void)in_sizes; (void)n_in; (void)out_size; (void)ws_size;
}
